// CrossAttention_26697516712511
// MI455X (gfx1250) — matrix operand order repaired and checked by reading, not compiled, not run on hardware
//
#include <hip/hip_runtime.h>

// ---------------------------------------------------------------------------
// Cross-attention (B=8, C=256, d=32, H=W=64 -> N=4096) for MI455X (gfx1250).
// Flash-style two-pass softmax-over-q attention, all matmuls on
// v_wmma_f32_16x16x32_bf16. ~100 MB mandatory HBM traffic, attention matrix
// (537 MB in fp32) is never materialized.
// ---------------------------------------------------------------------------

#define NBATCH 8
#define NCH    256
#define NDIM   32
#define NPIX   4096   // 64*64

typedef __attribute__((ext_vector_type(16))) __bf16        v16bf;
typedef __attribute__((ext_vector_type(2)))  __bf16        v2bf;
typedef __attribute__((ext_vector_type(8)))  float         v8f;
typedef __attribute__((ext_vector_type(4)))  unsigned int  u32x4;

union FragBF { v16bf v; u32x4 q[2]; };

// ---- workspace layout (elements of bf16) ----------------------------------
constexpr size_t OFF_WQ = 0;                       // 32x256  A-frags (8192)
constexpr size_t OFF_WK = 8192;
constexpr size_t OFF_WV = 16384;
constexpr size_t OFF_WO = 24576;                   // 256x32  A-frags (8192)
constexpr size_t OFF_QT = 32768;
constexpr size_t SZ_MAT = (size_t)NBATCH * NPIX * NDIM;   // 1,048,576 elems
constexpr size_t OFF_KT = OFF_QT + SZ_MAT;
constexpr size_t OFF_V  = OFF_KT + SZ_MAT;
constexpr size_t OFF_OT = OFF_V  + SZ_MAT;
constexpr size_t BF16_END = OFF_OT + SZ_MAT;              // elems
constexpr size_t M_BYTE_OFF = BF16_END * 2;               // bytes into ws

// float -> bf16: prefer packed hw convert, else native scalar convert
// (clang lowers fptrunc to v_cvt on targets with bf16 support).
__device__ __forceinline__ unsigned pack2_bf(float lo, float hi) {
#if __has_builtin(__builtin_amdgcn_cvt_pk_bf16_f32)
  v2bf p = __builtin_amdgcn_cvt_pk_bf16_f32(lo, hi);
  return __builtin_bit_cast(unsigned, p);
#else
  v2bf p = {static_cast<__bf16>(lo), static_cast<__bf16>(hi)};
  return __builtin_bit_cast(unsigned, p);
#endif
}

__device__ __forceinline__ unsigned short f2bf_bits(float f) {
  return (unsigned short)__builtin_bit_cast(unsigned short,
                                            static_cast<__bf16>(f));
}

__device__ __forceinline__ u32x4 pack_bf8(v8f a) {
  u32x4 r;
  unsigned* pu = (unsigned*)&r;
#pragma unroll
  for (int j = 0; j < 4; ++j) pu[j] = pack2_bf(a[2 * j], a[2 * j + 1]);
  return r;
}

__device__ __forceinline__ v8f wmma_bf16(v16bf a, v16bf b, v8f c) {
  return __builtin_amdgcn_wmma_f32_16x16x32_bf16(false, a, false, b,
                                                 (short)0, c, false, false);
}

// A-frag (16-bit 16x32): lane L holds row M=L%16; element e -> K =
// (L/16)*16 + e  => one contiguous 16-elem (32B) chunk.
__device__ __forceinline__ v16bf load_afrag(const unsigned short* row, int half) {
  FragBF f;
  f.q[0] = *(const u32x4*)(row + half * 16);
  f.q[1] = *(const u32x4*)(row + half * 16 + 8);
  return f.v;
}
// B-frag (16-bit 32x16): lane L holds col N=L%16; element e -> K = (L/16)*16+e
// => one contiguous 16-elem (32B) chunk.
__device__ __forceinline__ v16bf load_bfrag(const unsigned short* row, int half) {
  FragBF f;
  f.q[0] = *(const u32x4*)(row + half * 16);
  f.q[1] = *(const u32x4*)(row + half * 16 + 8);
  return f.v;
}

// ---------------------------------------------------------------------------
// K0: swizzle fp32 weights into bf16 WMMA A-fragment order (once, tiny).
// ---------------------------------------------------------------------------
__global__ __launch_bounds__(256) void prep_weights(
    const float* __restrict__ Wq, const float* __restrict__ Wk,
    const float* __restrict__ Wv, const float* __restrict__ Wo,
    unsigned short* __restrict__ ws) {
  // Wq/Wk/Wv: 32x256, slot = ((mt*8+kt)*32+lane)*16+e
  for (int idx = threadIdx.x; idx < 8192; idx += 256) {
    int e = idx & 15, lane = (idx >> 4) & 31, kt = (idx >> 9) & 7, mt = idx >> 12;
    int half = lane >> 4;
    int m = mt * 16 + (lane & 15);
    int kl = half * 16 + e;
    int c = kt * 32 + kl;
    ws[OFF_WQ + idx] = f2bf_bits(Wq[m * NCH + c]);
    ws[OFF_WK + idx] = f2bf_bits(Wk[m * NCH + c]);
    ws[OFF_WV + idx] = f2bf_bits(Wv[m * NCH + c]);
  }
  // Wo: 256x32 (K'=32, one k-tile), slot = (mt*32+lane)*16+e
  for (int idx = threadIdx.x; idx < 8192; idx += 256) {
    int e = idx & 15, lane = (idx >> 4) & 31, mt = idx >> 9;
    int half = lane >> 4;
    int m = mt * 16 + (lane & 15);
    int kl = half * 16 + e;
    ws[OFF_WO + idx] = f2bf_bits(Wo[m * NDIM + kl]);
  }
}

// ---------------------------------------------------------------------------
// K1: Q/K/V 1x1-conv projections. Block = 64 pixels of one batch; inputs are
// staged to LDS in B-fragment order (coalesced fp32 reads, packed bf16
// converts, dword LDS stores). Q,K stored transposed (N,32); V (32,N).
// ---------------------------------------------------------------------------
__global__ __launch_bounds__(256) void qkv_proj(
    const float* __restrict__ gaf, const float* __restrict__ mtf,
    const float* __restrict__ bq, const float* __restrict__ bk,
    const float* __restrict__ bv, unsigned short* __restrict__ ws) {
  __shared__ __align__(16) unsigned short ldsX[2][16384];  // 64 KB
  const int b  = blockIdx.y;
  const int n0 = blockIdx.x * 64;

  // Stage gaf / mtf tile (256ch x 64pix) as bf16 B-fragments. Channel pairs
  // (c, c+1) land in adjacent fragment slots -> packed convert + b32 store.
  for (int s = 0; s < 2; ++s) {
    const float* x = s ? mtf : gaf;
    for (int idx = threadIdx.x; idx < 8192; idx += 256) {
      int cp = idx >> 6, nl = idx & 63;                // coalesced in nl
      int c = cp * 2;                                  // even channel
      int kt = c >> 5, half = (c >> 4) & 1, e = c & 15;
      int ntile = nl >> 4, lane = half * 16 + (nl & 15);
      int slot = ((ntile * 8 + kt) * 32 + lane) * 16 + e;   // even
      size_t g = ((size_t)b * NCH + c) * NPIX + n0 + nl;
      *(unsigned*)&ldsX[s][slot] = pack2_bf(x[g], x[g + NPIX]);
    }
  }
  __syncthreads();

  const int wave = __builtin_amdgcn_readfirstlane(threadIdx.x >> 5);
  const int lane = threadIdx.x & 31;
  const int half = lane >> 4;

  for (int t = 0; t < 3; ++t) {                        // 3 proj x 2 mt x 4 nt
    const int task  = __builtin_amdgcn_readfirstlane(wave + t * 8);
    const int proj  = task >> 3;
    const int mt    = (task >> 2) & 1;
    const int ntile = task & 3;
    const unsigned short* wfrag =
        ws + (proj == 0 ? OFF_WQ : proj == 1 ? OFF_WK : OFF_WV) +
        (size_t)mt * 8 * 512;
    const float* bias = proj == 0 ? bq : proj == 1 ? bk : bv;

    v8f acc;
#pragma unroll
    for (int j = 0; j < 8; ++j) acc[j] = bias[mt * 16 + half * 8 + j];

    const unsigned short* xf = &ldsX[proj == 0 ? 0 : 1][(ntile * 8) * 512];
#pragma unroll
    for (int kt = 0; kt < 8; ++kt) {
      FragBF a, bfr;
      a.q[0]   = *(const u32x4*)(wfrag + kt * 512 + lane * 16);
      a.q[1]   = *(const u32x4*)(wfrag + kt * 512 + lane * 16 + 8);
      bfr.q[0] = *(const u32x4*)(xf + kt * 512 + lane * 16);
      bfr.q[1] = *(const u32x4*)(xf + kt * 512 + lane * 16 + 8);
      acc = wmma_bf16(a.v, bfr.v, acc);
    }

    const int n = n0 + ntile * 16 + (lane & 15);
    if (proj < 2) {  // Qt / Kt : (b, n, d) — 16B packed store per lane
      size_t base = (proj == 0 ? OFF_QT : OFF_KT) +
                    ((size_t)b * NPIX + n) * NDIM + mt * 16 + half * 8;
      *(u32x4*)(ws + base) = pack_bf8(acc);
    } else {         // V : (b, d, n) — bf16 scatter along d
#pragma unroll
      for (int j = 0; j < 8; ++j) {
        int m = mt * 16 + half * 8 + j;
        ws[OFF_V + ((size_t)b * NDIM + m) * NPIX + n] = f2bf_bits(acc[j]);
      }
    }
  }
}

// ---------------------------------------------------------------------------
// K2: per-(b,k) softmax stats over q: m = max_q S[k,q], Z = sum_q exp(S-m).
// One wave per 16-k tile, streaming S = K^T Q over q with WMMA, online
// max/sum, then cross-lane merge within each 16-lane half via shfl_xor.
// ---------------------------------------------------------------------------
__global__ __launch_bounds__(256) void row_stats(
    const unsigned short* __restrict__ ws, float* __restrict__ mOut,
    float* __restrict__ zOut) {
  const int lane = threadIdx.x & 31, half = lane >> 4;
  const int task =
      __builtin_amdgcn_readfirstlane(blockIdx.x * 8 + (threadIdx.x >> 5));
  const int b = task >> 8, kt = task & 255;

  const unsigned short* krow =
      ws + OFF_KT + ((size_t)b * NPIX + kt * 16 + (lane & 15)) * NDIM;
  v16bf afrag = load_afrag(krow, half);

  float mj[8], zj[8];
#pragma unroll
  for (int j = 0; j < 8; ++j) { mj[j] = -1e30f; zj[j] = 0.f; }

  const unsigned short* qbase =
      ws + OFF_QT + ((size_t)b * NPIX + (lane & 15)) * NDIM;
  for (int q0 = 0; q0 < NPIX; q0 += 16) {
    v16bf bfrag = load_bfrag(qbase + (size_t)q0 * NDIM, half);
    v8f zc = {};
    v8f s = wmma_bf16(afrag, bfrag, zc);
#pragma unroll
    for (int j = 0; j < 8; ++j) {
      float sv = s[j];
      float mn = fmaxf(mj[j], sv);
      zj[j] = zj[j] * __expf(mj[j] - mn) + __expf(sv - mn);
      mj[j] = mn;
    }
  }
  // merge the 16 q-columns held across lanes of each half-group
#pragma unroll
  for (int off = 1; off < 16; off <<= 1) {
#pragma unroll
    for (int j = 0; j < 8; ++j) {
      float m2 = __shfl_xor(mj[j], off, 16);
      float z2 = __shfl_xor(zj[j], off, 16);
      float mn = fmaxf(mj[j], m2);
      zj[j] = zj[j] * __expf(mj[j] - mn) + z2 * __expf(m2 - mn);
      mj[j] = mn;
    }
  }
  if ((lane & 15) == 0) {
#pragma unroll
    for (int j = 0; j < 8; ++j) {
      int k = kt * 16 + half * 8 + j;
      mOut[b * NPIX + k] = mj[j];
      zOut[b * NPIX + k] = zj[j];
    }
  }
}

// ---------------------------------------------------------------------------
// K3: out_pre[:,q] = sum_k V[:,k] * exp(S[k,q]-m_k)/Z_k, recomputing S tiles
// (avoids the 537 MB attention matrix). P is re-laid-out f32-acc -> bf16
// B-fragment through a per-wave LDS transpose buffer.
// ---------------------------------------------------------------------------
__global__ __launch_bounds__(256) void attn_apply(
    unsigned short* __restrict__ ws, const float* __restrict__ mIn,
    const float* __restrict__ zIn) {
  __shared__ __align__(16) unsigned short pbuf[8][512];  // [wave][q16 x k32]
  const int lane = threadIdx.x & 31, half = lane >> 4;
  const int wave = __builtin_amdgcn_readfirstlane(threadIdx.x >> 5);
  const int task = __builtin_amdgcn_readfirstlane(blockIdx.x * 8 + wave);
  const int b = task >> 8, qt = task & 255;

  const unsigned short* qrow =
      ws + OFF_QT + ((size_t)b * NPIX + qt * 16 + (lane & 15)) * NDIM;
  v16bf bq_ = load_bfrag(qrow, half);  // Q fragment, reused for all k

  v8f acc0 = {}, acc1 = {};
  for (int kc = 0; kc < NPIX; kc += 32) {
#pragma unroll
    for (int sub = 0; sub < 2; ++sub) {
      const int k0 = kc + sub * 16;
      const unsigned short* krow =
          ws + OFF_KT + ((size_t)b * NPIX + k0 + (lane & 15)) * NDIM;
      v16bf ak = load_afrag(krow, half);
      v8f zc = {};
      v8f s = wmma_bf16(ak, bq_, zc);                 // S tile 16k x 16q
      const float* mrow = mIn + b * NPIX + k0 + half * 8;
      const float* zrow = zIn + b * NPIX + k0 + half * 8;
      u32x4 pk;
      unsigned* pu = (unsigned*)&pk;
#pragma unroll
      for (int j = 0; j < 4; ++j) {
        float p0 = __expf(s[2 * j]     - mrow[2 * j])     / zrow[2 * j];
        float p1 = __expf(s[2 * j + 1] - mrow[2 * j + 1]) / zrow[2 * j + 1];
        pu[j] = pack2_bf(p0, p1);
      }
      // transpose-store: pbuf[q_local][k_local], 16B per lane
      *(u32x4*)&pbuf[wave][(lane & 15) * 32 + sub * 16 + half * 8] = pk;
    }
    asm volatile("s_wait_dscnt 0" ::: "memory");      // wave-private LDS RAW

    FragBF bp;
    bp.q[0] = *(const u32x4*)&pbuf[wave][(lane & 15) * 32 + half * 16];
    bp.q[1] = *(const u32x4*)&pbuf[wave][(lane & 15) * 32 + half * 16 + 8];

    const unsigned short* vrow0 =
        ws + OFF_V + ((size_t)b * NDIM + (lane & 15)) * NPIX + kc;
    const unsigned short* vrow1 =
        ws + OFF_V + ((size_t)b * NDIM + 16 + (lane & 15)) * NPIX + kc;
    if (kc + 32 < NPIX) __builtin_prefetch(vrow0 + 32, 0, 1);
    v16bf av0 = load_afrag(vrow0, half);
    v16bf av1 = load_afrag(vrow1, half);
    acc0 = wmma_bf16(av0, bp.v, acc0);
    acc1 = wmma_bf16(av1, bp.v, acc1);
  }

  // Ot[b][q][cd] bf16, packed 16B stores
  const int q = qt * 16 + (lane & 15);
  unsigned short* orow = ws + OFF_OT + ((size_t)b * NPIX + q) * NDIM;
  *(u32x4*)(orow + half * 8)      = pack_bf8(acc0);
  *(u32x4*)(orow + 16 + half * 8) = pack_bf8(acc1);
}

// ---------------------------------------------------------------------------
// K4: final 1x1 conv Wo (256x32) + bias + gamma*out + gaf residual.
// ---------------------------------------------------------------------------
__global__ __launch_bounds__(256) void out_proj(
    const unsigned short* __restrict__ ws, const float* __restrict__ gaf,
    const float* __restrict__ bo, const float* __restrict__ gamma,
    float* __restrict__ out) {
  const int lane = threadIdx.x & 31, half = lane >> 4;
  const int task =
      __builtin_amdgcn_readfirstlane(blockIdx.x * 8 + (threadIdx.x >> 5));
  const int b = task >> 8, qt = task & 255;
  const float g = gamma[0];

  const unsigned short* orow =
      ws + OFF_OT + ((size_t)b * NPIX + qt * 16 + (lane & 15)) * NDIM;
  v16bf bfrag = load_bfrag(orow, half);
  const int n = qt * 16 + (lane & 15);

  for (int mt = 0; mt < 16; ++mt) {
    FragBF a;
    const unsigned short* wf = ws + OFF_WO + (size_t)mt * 512 + lane * 16;
    a.q[0] = *(const u32x4*)wf;
    a.q[1] = *(const u32x4*)(wf + 8);
    v8f zc = {};
    v8f acc = wmma_bf16(a.v, bfrag, zc);
#pragma unroll
    for (int j = 0; j < 8; ++j) {
      int c = mt * 16 + half * 8 + j;
      size_t off = ((size_t)b * NCH + c) * NPIX + n;
      out[off] = g * (acc[j] + bo[c]) + gaf[off];
    }
  }
}

// ---------------------------------------------------------------------------
extern "C" void kernel_launch(void* const* d_in, const int* in_sizes, int n_in,
                              void* d_out, int out_size, void* d_ws, size_t ws_size,
                              hipStream_t stream) {
  (void)in_sizes; (void)n_in; (void)out_size; (void)ws_size;
  const float* gaf   = (const float*)d_in[0];
  const float* mtf   = (const float*)d_in[1];
  const float* Wq    = (const float*)d_in[2];
  const float* bq    = (const float*)d_in[3];
  const float* Wk    = (const float*)d_in[4];
  const float* bk    = (const float*)d_in[5];
  const float* Wv    = (const float*)d_in[6];
  const float* bv    = (const float*)d_in[7];
  const float* Wo    = (const float*)d_in[8];
  const float* bo    = (const float*)d_in[9];
  const float* gamma = (const float*)d_in[10];

  unsigned short* ws = (unsigned short*)d_ws;
  float* mbuf = (float*)((char*)d_ws + M_BYTE_OFF);
  float* zbuf = mbuf + NBATCH * NPIX;

  prep_weights<<<1, 256, 0, stream>>>(Wq, Wk, Wv, Wo, ws);
  qkv_proj<<<dim3(NPIX / 64, NBATCH), 256, 0, stream>>>(gaf, mtf, bq, bk, bv, ws);
  row_stats<<<NBATCH * NPIX / 16 / 8, 256, 0, stream>>>(ws, mbuf, zbuf);
  attn_apply<<<NBATCH * NPIX / 16 / 8, 256, 0, stream>>>(ws, mbuf, zbuf);
  out_proj<<<NBATCH * NPIX / 16 / 8, 256, 0, stream>>>(ws, gaf, bo, gamma,
                                                       (float*)d_out);
}
